// ParallelDense_34600256537436
// MI455X (gfx1250) — compile-verified, matrix operand order checked by reading, not run on hardware
//
#include <hip/hip_runtime.h>

// ---- CDNA5 WMMA vector types (probe-verified signatures) ----
typedef __attribute__((ext_vector_type(16))) __bf16 v16bf;
typedef __attribute__((ext_vector_type(8)))  float  v8f;

union FragU {
    v16bf v;
    uint4 q[2];
};

#define TB    256
#define BM    128
#define BN    128
#define BK    32
#define LDP   40          // LDS row pitch in bf16 elems; mult. of 8 keeps 16B chunks aligned, 20-bank row stride

#define MMASK 64
#define FFEAT 512
#define UUNIT 512

__global__ __launch_bounds__(TB)
void pdense_wmma_bf16x2(const float* __restrict__ X,
                        const float* __restrict__ Wt,
                        const float* __restrict__ Bias,
                        float* __restrict__ Y)
{
    __shared__ __align__(16) __bf16 AsH[BM * LDP];
    __shared__ __align__(16) __bf16 AsL[BM * LDP];
    __shared__ __align__(16) __bf16 BsH[BN * LDP];
    __shared__ __align__(16) __bf16 BsL[BN * LDP];

    const int m       = blockIdx.z;        // mask index
    const int rowBase = blockIdx.x * BM;   // batch rows
    const int colBase = blockIdx.y * BN;   // units
    const int tid     = threadIdx.x;
    const int lane    = tid & 31;
    const int wave    = tid >> 5;
    const int wm      = wave >> 2;         // 0..1 : 64-row slab
    const int wn      = wave & 3;          // 0..3 : 32-col slab

    const float* Wm = Wt + (size_t)m * FFEAT * UUNIT;

    // A staging: 128 rows x 32 k; 2 threads/row, 16 floats each
    const int a_row = tid >> 1;
    const int a_k   = (tid & 1) << 4;
    // B staging: 32 k-rows x 128 n; 8 threads/row, 16 floats each
    const int b_k   = tid >> 3;
    const int b_n   = (tid & 7) << 4;

    v8f zero = {};
    v8f acc[4][2];
    #pragma unroll
    for (int i = 0; i < 4; ++i)
        #pragma unroll
        for (int j = 0; j < 2; ++j)
            acc[i][j] = zero;

    // ISA 16-bit operand layout: lanes 0-15 hold K 0-7 / 16-23, lanes 16-31 hold K 8-15 / 24-31
    const int lrow = lane & 15;
    const int ksel = (lane >> 4) << 3;

    const float* aSrcBase = X  + (size_t)(rowBase + a_row) * (MMASK * FFEAT) + (size_t)m * FFEAT + a_k;
    const float* bSrcBase = Wm + (size_t)b_k * UUNIT + colBase + b_n;

    for (int kt = 0; kt < FFEAT; kt += BK) {
        // prefetch next k-tile (lowers to global_prefetch_b8; hides L2->WGP latency)
        if (kt + BK < FFEAT) {
            __builtin_prefetch(aSrcBase + kt + BK, 0, 3);
            __builtin_prefetch(bSrcBase + (size_t)(kt + BK) * UUNIT, 0, 3);
        }

        // ---- stage A tile: fp32 -> bf16 hi/lo split (native v_cvt bf16) ----
        const float4* ap = (const float4*)(aSrcBase + kt);
        #pragma unroll
        for (int c = 0; c < 4; ++c) {
            float4 f = ap[c];
            float fv[4] = {f.x, f.y, f.z, f.w};
            #pragma unroll
            for (int e = 0; e < 4; ++e) {
                __bf16 hi = (__bf16)fv[e];
                __bf16 lo = (__bf16)(fv[e] - (float)hi);
                int idx = a_row * LDP + a_k + c * 4 + e;
                AsH[idx] = hi;
                AsL[idx] = lo;
            }
        }
        // ---- stage B tile transposed into Bs[n][k] ----
        const float4* bp = (const float4*)(bSrcBase + (size_t)kt * UUNIT);
        #pragma unroll
        for (int c = 0; c < 4; ++c) {
            float4 f = bp[c];
            float fv[4] = {f.x, f.y, f.z, f.w};
            #pragma unroll
            for (int e = 0; e < 4; ++e) {
                __bf16 hi = (__bf16)fv[e];
                __bf16 lo = (__bf16)(fv[e] - (float)hi);
                int n = b_n + c * 4 + e;
                BsH[n * LDP + b_k] = hi;
                BsL[n * LDP + b_k] = lo;
            }
        }
        __syncthreads();

        // ---- fragment loads (two 16B chunks per operand half) ----
        FragU ah[4], al[4], bh[2], bl[2];
        #pragma unroll
        for (int i = 0; i < 4; ++i) {
            const __bf16* ph = &AsH[(wm * 64 + i * 16 + lrow) * LDP];
            const __bf16* pl = &AsL[(wm * 64 + i * 16 + lrow) * LDP];
            ah[i].q[0] = *(const uint4*)(ph + ksel);
            ah[i].q[1] = *(const uint4*)(ph + 16 + ksel);
            al[i].q[0] = *(const uint4*)(pl + ksel);
            al[i].q[1] = *(const uint4*)(pl + 16 + ksel);
        }
        #pragma unroll
        for (int j = 0; j < 2; ++j) {
            const __bf16* ph = &BsH[(wn * 32 + j * 16 + lrow) * LDP];
            const __bf16* pl = &BsL[(wn * 32 + j * 16 + lrow) * LDP];
            bh[j].q[0] = *(const uint4*)(ph + ksel);
            bh[j].q[1] = *(const uint4*)(ph + 16 + ksel);
            bl[j].q[0] = *(const uint4*)(pl + ksel);
            bl[j].q[1] = *(const uint4*)(pl + 16 + ksel);
        }

        // ---- split-precision product: hi*hi + hi*lo + lo*hi (lo*lo ~ 2^-16, dropped) ----
        #pragma unroll
        for (int i = 0; i < 4; ++i) {
            #pragma unroll
            for (int j = 0; j < 2; ++j) {
                acc[i][j] = __builtin_amdgcn_wmma_f32_16x16x32_bf16(
                    false, ah[i].v, false, bh[j].v, (short)0, acc[i][j], false, false);
                acc[i][j] = __builtin_amdgcn_wmma_f32_16x16x32_bf16(
                    false, ah[i].v, false, bl[j].v, (short)0, acc[i][j], false, false);
                acc[i][j] = __builtin_amdgcn_wmma_f32_16x16x32_bf16(
                    false, al[i].v, false, bh[j].v, (short)0, acc[i][j], false, false);
            }
        }
        __syncthreads();
    }

    // ---- epilogue: bias + relu + store ----
    // C/D layout: lanes 0-15 -> N=lane, M=vgpr; lanes 16-31 -> N=lane-16, M=8+vgpr
    const int ncol = lane & 15;
    const int mro  = (lane >> 4) << 3;
    #pragma unroll
    for (int j = 0; j < 2; ++j) {
        int col  = colBase + wn * 32 + j * 16 + ncol;
        float bv = Bias[m * UUNIT + col];
        #pragma unroll
        for (int i = 0; i < 4; ++i) {
            #pragma unroll
            for (int e = 0; e < 8; ++e) {
                int row = rowBase + wm * 64 + i * 16 + mro + e;
                float v = acc[i][j][e] + bv;
                Y[(size_t)row * (MMASK * UUNIT) + (size_t)m * UUNIT + col] = (v > 0.0f) ? v : 0.0f;
            }
        }
    }
}

extern "C" void kernel_launch(void* const* d_in, const int* in_sizes, int n_in,
                              void* d_out, int out_size, void* d_ws, size_t ws_size,
                              hipStream_t stream) {
    (void)n_in; (void)out_size; (void)d_ws; (void)ws_size;
    const float* x    = (const float*)d_in[0];   // [B, 64, 512] fp32
    const float* W    = (const float*)d_in[1];   // [64, 512, 512] fp32
    const float* bias = (const float*)d_in[2];   // [64, 512] fp32
    float* out        = (float*)d_out;           // [B, 64*512] fp32

    const int Bb = in_sizes[0] / (MMASK * FFEAT);   // 2048
    dim3 grid(Bb / BM, UUNIT / BN, MMASK);          // (16, 4, 64)
    dim3 block(TB);
    hipLaunchKernelGGL(pdense_wmma_bf16x2, grid, block, 0, stream, x, W, bias, out);
}
